// HalfKPModel_42958262895064
// MI455X (gfx1250) — matrix operand order repaired and checked by reading, not compiled, not run on hardware
//
#include <hip/hip_runtime.h>
#include <hip/hip_bf16.h>

typedef _Float16 v16h __attribute__((ext_vector_type(16)));
typedef _Float16 v8h  __attribute__((ext_vector_type(8)));
typedef _Float16 v4h  __attribute__((ext_vector_type(4)));
typedef float    v8f  __attribute__((ext_vector_type(8)));
typedef float    v4f  __attribute__((ext_vector_type(4)));

typedef unsigned int ui32x4 __attribute__((ext_vector_type(4)));
typedef int          i32x8  __attribute__((ext_vector_type(8)));
typedef int          i32x4  __attribute__((ext_vector_type(4)));

#define INPUT_DIM 40960
#define HIDDEN    256
#define BATCH     4096
#define M_TILE    32
#define K_TILE    64
#define B_STRIDE  80   // 64 data halves + 16 pad halves -> 160B rows (32B-aligned)
#define A_STRIDE  80

// ---------------------------------------------------------------------------
// Kernel 1: convert feat_w [256 x 40960] f32 -> f16 into workspace (once).
// ---------------------------------------------------------------------------
__global__ __launch_bounds__(256)
void halfkp_cvt_w(const float* __restrict__ w, _Float16* __restrict__ wh, int n) {
    int i = (blockIdx.x * blockDim.x + threadIdx.x) * 4;
    if (i < n) {
        v4f f = *(const v4f*)(w + i);
        v4h h;
        h[0] = (_Float16)f[0];
        h[1] = (_Float16)f[1];
        h[2] = (_Float16)f[2];
        h[3] = (_Float16)f[3];
        *(v4h*)(wh + i) = h;
    }
}

// ---------------------------------------------------------------------------
// TDM: issue one tensor_load_to_lds moving the whole 256x64 f16 B tile
// (32 KB) from global (L2-resident f16 weights) into padded LDS rows.
// D# per CDNA5 ISA ch.8: 2D tile, data_size=2B, pad 8 DWORDs after every
// 32 DWORDs (-> 160B LDS row stride = B_STRIDE halves).
// ---------------------------------------------------------------------------
__device__ __forceinline__
void tdm_load_b_tile(const _Float16* wh, int k0, unsigned lds_addr) {
    unsigned long long ga = (unsigned long long)(uintptr_t)(wh + k0);

    ui32x4 g0;
    g0[0] = 1u;                                    // count=1, is_restore=0
    g0[1] = lds_addr;                              // LDS byte address of tile
    g0[2] = (unsigned)ga;                          // global_addr[31:0]
    g0[3] = (unsigned)(ga >> 32) | (2u << 30);     // global_addr[56:32] | type=2

    i32x8 g1;
    g1[0] = (1 << 16)                              // data_size = 2 bytes
          | (1 << 20)                              // pad_enable
          | (4 << 22)                              // pad_interval: every 32 DWORDs
          | (7 << 25);                             // pad_amount: 8 DWORDs (32B)
    g1[1] = (int)(0xA000u << 16);                  // tensor_dim0 = 40960 (lo16)
    g1[2] = (int)(0x0100u << 16);                  // tensor_dim0 hi=0 | tensor_dim1=256 (lo16)
    g1[3] = (64 << 16);                            // tensor_dim1 hi=0 | tile_dim0=64
    g1[4] = 256;                                   // tile_dim1=256 | tile_dim2=0
    g1[5] = 40960;                                 // tensor_dim0_stride (lo32)
    g1[6] = 0;                                     // stride hi | tensor_dim1_stride lo
    g1[7] = 0;

    i32x4 gz = {0, 0, 0, 0};                       // 2D tensor: groups 2/3 unused
#if __clang_major__ >= 23
    i32x8 gz8 = {0, 0, 0, 0, 0, 0, 0, 0};
    __builtin_amdgcn_tensor_load_to_lds(g0, g1, gz, gz, gz8, 0);
#else
    __builtin_amdgcn_tensor_load_to_lds(g0, g1, gz, gz, 0);
#endif
}

// ---------------------------------------------------------------------------
// Kernel 2: fused feature-layer GEMM + clipped ReLU + out_w dot partials.
// Block: 256 threads = 8 wave32, tile M=32 x N=256, K-step 64, double-
// buffered LDS; B tiles arrive via TDM (overlapped), A tiles via VALU cvt.
// ---------------------------------------------------------------------------
__global__ __launch_bounds__(256)
void halfkp_gemm(const float*   __restrict__ wf,
                 const float*   __restrict__ bfeat,
                 const _Float16* __restrict__ wh,
                 const float*   __restrict__ feat_b,
                 const float*   __restrict__ out_w,
                 float*         __restrict__ partials) {
    __shared__ _Float16 As[2][M_TILE * A_STRIDE];   // 2 x 5 KB
    __shared__ _Float16 Bs[2][HIDDEN * B_STRIDE];   // 2 x 40 KB
    __shared__ float    rowpart[M_TILE];

    const int side = blockIdx.y;
    const float* feat = side ? bfeat : wf;
    const int row0 = blockIdx.x * M_TILE;

    const int tid   = threadIdx.x;
    const int wave  = tid >> 5;            // 0..7
    const int lane  = tid & 31;
    const int m_sub = wave & 1;            // M offset 0 / 16
    const int n_base = (wave >> 1) * 64;   // N offset 0/64/128/192

    if (tid < M_TILE) rowpart[tid] = 0.0f;

    v8f acc[4] = {};                       // 4 x (16x16 f32) fragments

    // Per-lane WMMA fragment addressing (contiguous k = 16*lhalf + s in A and B).
    const int lhalf = lane >> 4;
    const int l15   = lane & 15;
    const int a_row = m_sub * 16 + l15;

    // A staging map: thread -> (m = tid/8, 8-float chunk kc = (tid&7)*8)
    const int st_m  = tid >> 3;
    const int st_kc = (tid & 7) * 8;
    const float* feat_row = feat + (size_t)(row0 + st_m) * INPUT_DIM + st_kc;

    const unsigned bs_lds[2] = { (unsigned)(uintptr_t)&Bs[0][0],
                                 (unsigned)(uintptr_t)&Bs[1][0] };

    // Prologue: wave 0 kicks off the TDM fetch of B tile 0.
    if (wave == 0) tdm_load_b_tile(wh, 0, bs_lds[0]);

    const int NK = INPUT_DIM / K_TILE;     // 640
    for (int i = 0; i < NK; ++i) {
        const int k0  = i * K_TILE;
        const int cur = i & 1;

        // ---- stage A tile: 32x64 f32 features -> f16 LDS (VALU cvt) ----
        {
            const float* src = feat_row + k0;
            v4f f0 = *(const v4f*)(src);
            v4f f1 = *(const v4f*)(src + 4);
            v8h h;
            h[0] = (_Float16)f0[0]; h[1] = (_Float16)f0[1];
            h[2] = (_Float16)f0[2]; h[3] = (_Float16)f0[3];
            h[4] = (_Float16)f1[0]; h[5] = (_Float16)f1[1];
            h[6] = (_Float16)f1[2]; h[7] = (_Float16)f1[3];
            *(v8h*)&As[cur][st_m * A_STRIDE + st_kc] = h;
            if (k0 + K_TILE < INPUT_DIM)
                __builtin_prefetch(src + K_TILE, 0, 1);     // global_prefetch_b8
        }

        // ---- TDM double-buffer: issue next B tile, wait for current ----
        if (wave == 0) {
            if (i + 1 < NK) {
                tdm_load_b_tile(wh, k0 + K_TILE, bs_lds[cur ^ 1]);
                __builtin_amdgcn_s_wait_tensorcnt((short)1);   // tile i resident
            } else {
                __builtin_amdgcn_s_wait_tensorcnt((short)0);
            }
        }
        __syncthreads();

        // ---- WMMA on tile i ----
        #pragma unroll
        for (int kk = 0; kk < K_TILE; kk += 32) {
            const int ak = kk + lhalf * 16;
            v8h a0 = *(const v8h*)&As[cur][a_row * A_STRIDE + ak];
            v8h a1 = *(const v8h*)&As[cur][a_row * A_STRIDE + ak + 8];
            v16h av = __builtin_shufflevector(a0, a1,
                        0,1,2,3,4,5,6,7,8,9,10,11,12,13,14,15);
            const int bk = kk + lhalf * 16;
            #pragma unroll
            for (int t = 0; t < 4; ++t) {
                const int bn = n_base + t * 16 + l15;
                v16h bv = *(const v16h*)&Bs[cur][bn * B_STRIDE + bk];
                acc[t] = __builtin_amdgcn_wmma_f32_16x16x32_f16(
                            /*neg_a=*/false, av, /*neg_b=*/false, bv,
                            /*c_mod=*/(short)0, acc[t],
                            /*reuse_a=*/false, /*reuse_b=*/false);
            }
        }
        __syncthreads();   // all waves done with buffer `cur` before overwrite
    }

    // ---- epilogue: bias + clamp(0,1), scale by out_w, reduce over N ----
    #pragma unroll
    for (int t = 0; t < 4; ++t) {
        const int n_g  = n_base + t * 16 + l15;
        const float bias = feat_b[n_g];
        const float ow   = out_w[side * HIDDEN + n_g];
        #pragma unroll
        for (int r = 0; r < 8; ++r) {
            float v = acc[t][r] + bias;
            v = fminf(fmaxf(v, 0.0f), 1.0f);
            float s = v * ow;
            s += __shfl_xor(s, 1);
            s += __shfl_xor(s, 2);
            s += __shfl_xor(s, 4);
            s += __shfl_xor(s, 8);
            if (l15 == 0) {
                const int m = m_sub * 16 + r + lhalf * 8;
                atomicAdd(&rowpart[m], s);     // ds_add_f32
            }
        }
    }
    __syncthreads();

    if (tid < M_TILE)
        partials[side * BATCH + row0 + tid] = rowpart[tid];
}

// ---------------------------------------------------------------------------
// Kernel 3: out[b] = white_partial[b] + black_partial[b] + out_b
// ---------------------------------------------------------------------------
__global__ __launch_bounds__(256)
void halfkp_out(const float* __restrict__ partials,
                const float* __restrict__ out_b,
                float*       __restrict__ out) {
    int i = blockIdx.x * blockDim.x + threadIdx.x;
    if (i < BATCH)
        out[i] = partials[i] + partials[BATCH + i] + out_b[0];
}

extern "C" void kernel_launch(void* const* d_in, const int* in_sizes, int n_in,
                              void* d_out, int out_size, void* d_ws, size_t ws_size,
                              hipStream_t stream) {
    const float* wf     = (const float*)d_in[0];
    const float* bfeat  = (const float*)d_in[1];
    const float* feat_w = (const float*)d_in[2];
    const float* feat_b = (const float*)d_in[3];
    const float* out_w  = (const float*)d_in[4];
    const float* out_b  = (const float*)d_in[5];
    float* out = (float*)d_out;

    // Workspace layout: [f16 feat_w : 21 MB][partials : 2*4096 f32]
    _Float16* wh = (_Float16*)d_ws;
    float* partials = (float*)((char*)d_ws + (size_t)HIDDEN * INPUT_DIM * sizeof(_Float16));

    const int nw = HIDDEN * INPUT_DIM;
    halfkp_cvt_w<<<nw / (256 * 4), 256, 0, stream>>>(feat_w, wh, nw);

    dim3 grid(BATCH / M_TILE, 2);
    halfkp_gemm<<<grid, 256, 0, stream>>>(wf, bfeat, wh, feat_b, out_w, partials);

    halfkp_out<<<BATCH / 256, 256, 0, stream>>>(partials, out_b, out);
}
